// Transformer_17231408791602
// MI455X (gfx1250) — matrix operand order repaired and checked by reading, not compiled, not run on hardware
//
#include <hip/hip_runtime.h>
#include <hip/hip_bf16.h>
#include <math.h>

typedef __bf16 bf16;
typedef bf16 v16bf __attribute__((ext_vector_type(16)));
typedef bf16 v8bf  __attribute__((ext_vector_type(8)));
typedef float v8f  __attribute__((ext_vector_type(8)));

// ---------------------------------------------------------------------------
// WMMA helpers (CDNA5 v_wmma_f32_16x16x32_bf16, wave32)
// ---------------------------------------------------------------------------
__device__ __forceinline__ v8f wmma_bf16(v16bf a, v16bf b, v8f c) {
  return __builtin_amdgcn_wmma_f32_16x16x32_bf16(
      /*neg_a=*/false, a, /*neg_b=*/false, b,
      /*c_mod=*/(short)0, c, /*reuse_a=*/false, /*reuse_b=*/false);
}

// A-matrix 16x32 bf16 fragment from row-major storage (pitch in elements).
// ISA layout: lane L holds row m=L%16; elements 0..7 = K[kb..kb+7],
// elements 8..15 = K[kb+16..kb+23], kb = (L<16 ? 0 : 8).
__device__ __forceinline__ v16bf load_a_frag(const bf16* base, int pitch) {
  const int lane = threadIdx.x & 31;
  const bf16* p = base + (size_t)(lane & 15) * pitch + ((lane >> 4) << 3);
  union { v16bf v; v8bf h[2]; } u;
  u.h[0] = *(const v8bf*)p;
  u.h[1] = *(const v8bf*)(p + 16);
  return u.v;
}

// B-matrix 32x16 bf16 fragment from COLUMN-major storage ([n][k], pitch in
// elements). Lane L holds column n=L%16; elements 0..7 = K[kb..kb+7],
// elements 8..15 = K[kb+16..kb+23], kb = (L<16 ? 0 : 8).
__device__ __forceinline__ v16bf load_b_frag(const bf16* base, int pitch) {
  const int lane = threadIdx.x & 31;
  const bf16* p = base + (size_t)(lane & 15) * pitch + ((lane >> 4) << 3);
  union { v16bf v; v8bf h[2]; } u;
  u.h[0] = *(const v8bf*)p;
  u.h[1] = *(const v8bf*)(p + 16);
  return u.v;
}

// CDNA5 async global->LDS DMA (ASYNCcnt-tracked). Generic LDS pointer's low
// 32 bits are the LDS byte offset (flat shared aperture lives in addr[63:32]).
__device__ __forceinline__ void async_b128(const bf16* g, bf16* l) {
  unsigned la = (unsigned)(uintptr_t)l;
  asm volatile("global_load_async_to_lds_b128 %0, %1, off"
               :: "v"(la), "v"(g) : "memory");
}

// ---------------------------------------------------------------------------
// fp32 -> bf16 conversion (plain, and transposing for weights)
// ---------------------------------------------------------------------------
__global__ void cvt_kernel(const float* __restrict__ src, bf16* __restrict__ dst, int n) {
  int i = blockIdx.x * 256 + threadIdx.x;
  if (i < n) dst[i] = (bf16)src[i];
}

// src is [K,N] row-major fp32; dst is [N,K] row-major bf16 (pre-transposed
// weights so GEMM B-tiles are K-contiguous and async-DMA friendly).
__global__ void cvt_t_kernel(const float* __restrict__ src, bf16* __restrict__ dst,
                             int K, int N) {
  int i = blockIdx.x * 256 + threadIdx.x;
  if (i < K * N) {
    int k = i / N, n = i - k * N;
    dst[(size_t)n * K + k] = (bf16)src[i];
  }
}

// ---------------------------------------------------------------------------
// LayerNorm over D=768, strided fp32 source, bf16 dest
// ---------------------------------------------------------------------------
__global__ __launch_bounds__(256)
void ln_kernel(const float* __restrict__ src, int stride, int off,
               const float* __restrict__ g, const float* __restrict__ bb,
               bf16* __restrict__ dst) {
  const int row = blockIdx.x, t = threadIdx.x;
  __shared__ float red[256];
  __shared__ float s_mu, s_rstd;
  const float* p = src + (size_t)row * 768 * stride + off;
  float v[3];
#pragma unroll
  for (int i = 0; i < 3; ++i) v[i] = p[(size_t)(t + i * 256) * stride];
  red[t] = v[0] + v[1] + v[2];
  __syncthreads();
  for (int o = 128; o > 0; o >>= 1) { if (t < o) red[t] += red[t + o]; __syncthreads(); }
  if (t == 0) s_mu = red[0] * (1.f / 768.f);
  __syncthreads();
  const float mu = s_mu;
  float d0 = v[0] - mu, d1 = v[1] - mu, d2 = v[2] - mu;
  red[t] = d0 * d0 + d1 * d1 + d2 * d2;
  __syncthreads();
  for (int o = 128; o > 0; o >>= 1) { if (t < o) red[t] += red[t + o]; __syncthreads(); }
  if (t == 0) s_rstd = rsqrtf(red[0] * (1.f / 768.f) + 1e-6f);
  __syncthreads();
  const float rstd = s_rstd;
#pragma unroll
  for (int i = 0; i < 3; ++i) {
    int d = t + i * 256;
    dst[(size_t)row * 768 + d] = (bf16)((v[i] - mu) * rstd * g[d] + bb[d]);
  }
}

// ---------------------------------------------------------------------------
// Tiled WMMA GEMM: C[M,N] = A[M,K](bf16) x Wt[N,K](bf16)^T + bias.
// 128x128 tile / workgroup, 8 waves, each wave 64x32 (4x2 WMMA subtiles).
// Double-buffered LDS fed by global_load_async_to_lds_b128 (ASYNCcnt).
// MODE 0: bf16 out          MODE 1: +res(f32,strided) -> outf & out[...,1]
// MODE 2: gelu -> bf16 out  MODE 3: +res(f32,strided) -> out[...,0]
// ---------------------------------------------------------------------------
template<int MODE>
__global__ __launch_bounds__(256)
void gemm_kernel(const bf16* __restrict__ A, const bf16* __restrict__ Wt,
                 const float* __restrict__ bias,
                 bf16* __restrict__ outb, float* __restrict__ outf,
                 float* __restrict__ outf2,
                 const float* __restrict__ res, int res_stride, int res_off,
                 int M, int N, int K) {
  __shared__ bf16 As[2][128 * 40];   // row-major A tiles, pitch 40 (pad)
  __shared__ bf16 Bs[2][128 * 40];   // row-major Wt tiles ([n][k]), pitch 40

  const int t = threadIdx.x, lane = t & 31, w = t >> 5;
  const int wm = w >> 2, wn = w & 3;
  const int m0 = blockIdx.y * 128;
  const int n0 = blockIdx.x * 128;

  // Per-thread 16B chunk coordinates: 512 chunks cover a 128x32 tile.
  const int id0 = t * 2, id1 = t * 2 + 1;
  const int r0 = id0 >> 2, c0 = (id0 & 3) * 8;
  const int r1 = id1 >> 2, c1 = (id1 & 3) * 8;

  const bf16* Arow0 = A + (size_t)(m0 + r0) * K + c0;
  const bf16* Arow1 = A + (size_t)(m0 + r1) * K + c1;
  const bf16* Brow0 = Wt + (size_t)(n0 + r0) * K + c0;
  const bf16* Brow1 = Wt + (size_t)(n0 + r1) * K + c1;
  bf16* la0;
  bf16* la1;
  bf16* lb0;
  bf16* lb1;

  v8f acc[4][2] = {};
  const int nk = K / 32;

  // Prologue: async-load tile 0 into buffer 0.
  la0 = &As[0][r0 * 40 + c0]; la1 = &As[0][r1 * 40 + c1];
  lb0 = &Bs[0][r0 * 40 + c0]; lb1 = &Bs[0][r1 * 40 + c1];
  async_b128(Arow0, la0); async_b128(Arow1, la1);
  async_b128(Brow0, lb0); async_b128(Brow1, lb1);

  for (int kt = 0; kt < nk; ++kt) {
    if (kt + 1 < nk) {
      const int nb = (kt + 1) & 1;
      const int k0 = (kt + 1) * 32;
      async_b128(Arow0 + k0, &As[nb][r0 * 40 + c0]);
      async_b128(Arow1 + k0, &As[nb][r1 * 40 + c1]);
      async_b128(Brow0 + k0, &Bs[nb][r0 * 40 + c0]);
      async_b128(Brow1 + k0, &Bs[nb][r1 * 40 + c1]);
      asm volatile("s_wait_asynccnt 4" ::: "memory");  // tile kt landed
    } else {
      asm volatile("s_wait_asynccnt 0" ::: "memory");
    }
    __syncthreads();  // all waves' DMA for tile kt visible

    const int cb = kt & 1;
    v16bf af[4], bfr[2];
#pragma unroll
    for (int i = 0; i < 4; ++i) af[i] = load_a_frag(&As[cb][(wm * 64 + i * 16) * 40], 40);
#pragma unroll
    for (int j = 0; j < 2; ++j) bfr[j] = load_b_frag(&Bs[cb][(wn * 32 + j * 16) * 40], 40);
#pragma unroll
    for (int i = 0; i < 4; ++i)
#pragma unroll
      for (int j = 0; j < 2; ++j) acc[i][j] = wmma_bf16(af[i], bfr[j], acc[i][j]);
    __syncthreads();  // buffer cb free for the DMA issued next iteration
  }

  const int rb = 8 * (lane >> 4);
#pragma unroll
  for (int i = 0; i < 4; ++i) {
#pragma unroll
    for (int j = 0; j < 2; ++j) {
      const int col = n0 + wn * 32 + j * 16 + (lane & 15);
      const float bcol = bias[col];
#pragma unroll
      for (int r = 0; r < 8; ++r) {
        const int row = m0 + wm * 64 + i * 16 + rb + r;
        float v = acc[i][j][r] + bcol;
        const size_t oi = (size_t)row * N + col;
        if (MODE == 0) {
          outb[oi] = (bf16)v;
        } else if (MODE == 1) {
          v += res[oi * res_stride + res_off];
          outf[oi] = v;
          outf2[oi * 2 + 1] = v;
        } else if (MODE == 2) {
          float x3 = v * v * v;
          float gl = 0.5f * v * (1.f + tanhf(0.7978845608028654f * (v + 0.044715f * x3)));
          outb[oi] = (bf16)gl;
        } else {
          v += res[oi * res_stride + res_off];
          outf2[oi * 2 + 0] = v;
        }
      }
    }
  }
}

// ---------------------------------------------------------------------------
// BigBird block-sparse attention: one workgroup per (b, h, query-block).
// Middle blocks: 1 chunk of 8 key-blocks {0, 63, win3, rand3}.
// Global blocks (0, 63): 8 chunks covering all 64 key-blocks, online softmax.
// ---------------------------------------------------------------------------
__global__ __launch_bounds__(256)
void attn_kernel(const bf16* __restrict__ qg, const bf16* __restrict__ kg,
                 const bf16* __restrict__ vg, const int* __restrict__ rand_attn,
                 bf16* __restrict__ ao) {
  constexpr int Sd = 4096, Dd = 768, NB = 64, R = 3, Hh = 12;
  const int idx = blockIdx.x;
  const int qb = idx % NB;
  const int h  = (idx / NB) % Hh;
  const int b  = idx / (NB * Hh);

  __shared__ bf16  Qs[64 * 72];          // Q block, pitch 72
  __shared__ float ScS[64 * 516];        // scores (f32); P(bf16) aliases row front
  __shared__ bf16  Vt[64 * 520];         // V transposed: Vt[hd][key], pitch 520
  __shared__ float mrow[64], lrow[64], arow[64];
  __shared__ int   kb_list[64];
  __shared__ int   nchunks_s;

  const int t = threadIdx.x;
  const int lane = t & 31;
  const int w = t >> 5;

  if (t == 0) {
    if (qb == 0 || qb == NB - 1) {
      nchunks_s = NB / 8;
      for (int i = 0; i < NB; ++i) kb_list[i] = i;
    } else {
      nchunks_s = 1;
      kb_list[0] = 0; kb_list[1] = NB - 1;
      kb_list[2] = qb - 1; kb_list[3] = qb; kb_list[4] = qb + 1;
      for (int r = 0; r < R; ++r)
        kb_list[5 + r] = rand_attn[(h * (NB - 2) + (qb - 1)) * R + r];
    }
  }
  if (t < 64) { mrow[t] = -INFINITY; lrow[t] = 0.f; }

  const size_t head_off = (size_t)h * 64;
  const bf16* qbase = qg + ((size_t)b * Sd + (size_t)qb * 64) * Dd + head_off;
  {  // stage Q block 64x64
    int r0 = t >> 2;
    int c0 = (t & 3) * 16;
    const bf16* src = qbase + (size_t)r0 * Dd + c0;
    *(v8bf*)&Qs[r0 * 72 + c0]     = *(const v8bf*)src;
    *(v8bf*)&Qs[r0 * 72 + c0 + 8] = *(const v8bf*)(src + 8);
  }
  __syncthreads();

  const int nchunks = nchunks_s;
  const bf16* kbase = kg + (size_t)b * Sd * Dd + head_off;
  const bf16* vbase = vg + (size_t)b * Sd * Dd + head_off;
  const float scale = 0.125f;  // 1/sqrt(64)

  const int om = w & 3;   // this wave's O rows  [16*om, +16)
  const int oc = w >> 2;  // this wave's O cols  [32*oc, +32)
  v8f oacc[2] = {};

  for (int ch = 0; ch < nchunks; ++ch) {
    // ---- stage V chunk transposed: Vt[hd][key], 512 gathered key rows ----
#pragma unroll
    for (int i = 0; i < 2; ++i) {
      int c = t * 2 + i;                         // key within chunk 0..511
      int blk = kb_list[ch * 8 + (c >> 6)];
      const bf16* vsrc = vbase + ((size_t)blk * 64 + (c & 63)) * Dd;
#pragma unroll
      for (int hc = 0; hc < 8; ++hc) {
        v8bf vv = *(const v8bf*)(vsrc + hc * 8);
#pragma unroll
        for (int e = 0; e < 8; ++e) Vt[(hc * 8 + e) * 520 + c] = vv[e];
      }
    }
    // ---- scores: wave w handles key block kb_list[ch*8+w] (cols w*64..) ----
    {
      const int blk = kb_list[ch * 8 + w];
      const bf16* kb0 = kbase + (size_t)blk * 64 * Dd;
#pragma unroll
      for (int j = 0; j < 4; ++j) {
        v8f acc[4] = {};
#pragma unroll
        for (int ks = 0; ks < 2; ++ks) {
          // B frag straight from global K rows (contiguous along HD)
          v16bf bfr = load_b_frag(kb0 + (size_t)(j * 16) * Dd + ks * 32, Dd);
#pragma unroll
          for (int i = 0; i < 4; ++i) {
            v16bf af = load_a_frag(&Qs[(i * 16) * 72 + ks * 32], 72);
            acc[i] = wmma_bf16(af, bfr, acc[i]);
          }
        }
        const int colc = w * 64 + j * 16 + (lane & 15);
        const int rb = 8 * (lane >> 4);
#pragma unroll
        for (int i = 0; i < 4; ++i)
#pragma unroll
          for (int r = 0; r < 8; ++r)
            ScS[(i * 16 + rb + r) * 516 + colc] = acc[i][r] * scale;
      }
    }
    __syncthreads();
    // ---- online softmax, one thread per query row ----
    if (t < 64) {
      float* srow = &ScS[t * 516];
      bf16* prow = (bf16*)srow;   // safe: p[j] clobbers s[j/2], already consumed
      const float mOld = mrow[t];
      float mc = -INFINITY;
      for (int j2 = 0; j2 < 512; ++j2) mc = fmaxf(mc, srow[j2]);
      const float mNew = fmaxf(mOld, mc);
      const float alpha = __expf(mOld - mNew);
      float lsum = 0.f;
      for (int j2 = 0; j2 < 512; ++j2) {
        float p = __expf(srow[j2] - mNew);
        lsum += p;
        prow[j2] = (bf16)p;
      }
      mrow[t] = mNew;
      lrow[t] = lrow[t] * alpha + lsum;
      arow[t] = alpha;
    }
    __syncthreads();
    // ---- rescale O, accumulate O += P @ V ----
    {
      const int rb = 8 * (lane >> 4);
#pragma unroll
      for (int j = 0; j < 2; ++j)
#pragma unroll
        for (int r = 0; r < 8; ++r) oacc[j][r] *= arow[om * 16 + rb + r];
      const bf16* pbase = (const bf16*)&ScS[(om * 16) * 516];
#pragma unroll
      for (int ks = 0; ks < 16; ++ks) {
        v16bf af = load_a_frag(pbase + ks * 32, 1032);  // pitch 516 f32 = 1032 bf16
#pragma unroll
        for (int j = 0; j < 2; ++j) {
          v16bf bfr = load_b_frag(&Vt[(oc * 32 + j * 16) * 520 + ks * 32], 520);
          oacc[j] = wmma_bf16(af, bfr, oacc[j]);
        }
      }
    }
    __syncthreads();
  }

  // ---- finalize: divide by row sum, store bf16 ----
  {
    const int rb = 8 * (lane >> 4);
    bf16* aobase = ao + ((size_t)b * Sd + (size_t)qb * 64) * Dd + head_off;
#pragma unroll
    for (int j = 0; j < 2; ++j)
#pragma unroll
      for (int r = 0; r < 8; ++r) {
        const int row = om * 16 + rb + r;
        const float v = oacc[j][r] / lrow[row];
        aobase[(size_t)row * Dd + oc * 32 + j * 16 + (lane & 15)] = (bf16)v;
      }
  }
}

// ---------------------------------------------------------------------------
// Host orchestration
// ---------------------------------------------------------------------------
extern "C" void kernel_launch(void* const* d_in, const int* in_sizes, int n_in,
                              void* d_out, int out_size, void* d_ws, size_t ws_size,
                              hipStream_t stream) {
  (void)in_sizes; (void)n_in; (void)out_size; (void)ws_size;
  const float* x    = (const float*)d_in[0];
  const float* ln1g = (const float*)d_in[1];
  const float* ln1b = (const float*)d_in[2];
  const float* ln2g = (const float*)d_in[3];
  const float* ln2b = (const float*)d_in[4];
  const float* wq = (const float*)d_in[5];
  const float* bq = (const float*)d_in[6];
  const float* wk = (const float*)d_in[7];
  const float* bk = (const float*)d_in[8];
  const float* wv = (const float*)d_in[9];
  const float* bv = (const float*)d_in[10];
  const float* wo = (const float*)d_in[11];
  const float* bo = (const float*)d_in[12];
  const float* w1 = (const float*)d_in[13];
  const float* b1 = (const float*)d_in[14];
  const float* w2 = (const float*)d_in[15];
  const float* b2 = (const float*)d_in[16];
  const int* rand_attn = (const int*)d_in[17];
  float* out = (float*)d_out;

  char* ws = (char*)d_ws;
  size_t off = 0;
  auto alloc = [&](size_t bytes) -> char* {
    char* p = ws + off;
    off += (bytes + 255) & ~(size_t)255;
    return p;
  };
  constexpr size_t Mrows = 8192;  // B*S
  // Pre-transposed bf16 weights: wX_t is [N][K].
  bf16* wq_t = (bf16*)alloc(768 * 768 * 2);
  bf16* wk_t = (bf16*)alloc(768 * 768 * 2);
  bf16* wv_t = (bf16*)alloc(768 * 768 * 2);
  bf16* wo_t = (bf16*)alloc(768 * 768 * 2);
  bf16* w1_t = (bf16*)alloc(768 * 3072 * 2);
  bf16* w2_t = (bf16*)alloc(3072 * 768 * 2);
  bf16* xn_b = (bf16*)alloc(Mrows * 768 * 2);  // also reused as LN2 output
  bf16* q_b  = (bf16*)alloc(Mrows * 768 * 2);
  bf16* k_b  = (bf16*)alloc(Mrows * 768 * 2);
  bf16* v_b  = (bf16*)alloc(Mrows * 768 * 2);
  bf16* ao_b = (bf16*)alloc(Mrows * 768 * 2);
  float* y2f = (float*)alloc(Mrows * 768 * 4);
  bf16* h_b  = q_b;  // 8192x3072 bf16 aliases q/k/v/ao (dead by fc1)

  dim3 blk(256);
  int n = 768 * 768;
  cvt_t_kernel<<<(n + 255) / 256, blk, 0, stream>>>(wq, wq_t, 768, 768);
  cvt_t_kernel<<<(n + 255) / 256, blk, 0, stream>>>(wk, wk_t, 768, 768);
  cvt_t_kernel<<<(n + 255) / 256, blk, 0, stream>>>(wv, wv_t, 768, 768);
  cvt_t_kernel<<<(n + 255) / 256, blk, 0, stream>>>(wo, wo_t, 768, 768);
  n = 768 * 3072;
  cvt_t_kernel<<<(n + 255) / 256, blk, 0, stream>>>(w1, w1_t, 768, 3072);
  cvt_t_kernel<<<(n + 255) / 256, blk, 0, stream>>>(w2, w2_t, 3072, 768);

  // LN1 on x1 = x[...,1]
  ln_kernel<<<Mrows, blk, 0, stream>>>(x, 2, 1, ln1g, ln1b, xn_b);

  dim3 g768(768 / 128, Mrows / 128);  // (6, 64)
  gemm_kernel<0><<<g768, blk, 0, stream>>>(xn_b, wq_t, bq, q_b, nullptr, nullptr,
                                           nullptr, 0, 0, (int)Mrows, 768, 768);
  gemm_kernel<0><<<g768, blk, 0, stream>>>(xn_b, wk_t, bk, k_b, nullptr, nullptr,
                                           nullptr, 0, 0, (int)Mrows, 768, 768);
  gemm_kernel<0><<<g768, blk, 0, stream>>>(xn_b, wv_t, bv, v_b, nullptr, nullptr,
                                           nullptr, 0, 0, (int)Mrows, 768, 768);

  attn_kernel<<<2 * 12 * 64, blk, 0, stream>>>(q_b, k_b, v_b, rand_attn, ao_b);

  // y2 = x2 + ao @ wo + bo   (x2 = x[...,0]); writes ws y2f and out[...,1]
  gemm_kernel<1><<<g768, blk, 0, stream>>>(ao_b, wo_t, bo, nullptr, y2f, out,
                                           x, 2, 0, (int)Mrows, 768, 768);
  // LN2 on y2
  ln_kernel<<<Mrows, blk, 0, stream>>>(y2f, 1, 0, ln2g, ln2b, xn_b);
  // h = gelu(y2n @ w1 + b1)
  dim3 g3072(3072 / 128, Mrows / 128);
  gemm_kernel<2><<<g3072, blk, 0, stream>>>(xn_b, w1_t, b1, h_b, nullptr, nullptr,
                                            nullptr, 0, 0, (int)Mrows, 3072, 768);
  // y1 = x1 + h @ w2 + b2 -> out[...,0]
  gemm_kernel<3><<<g768, blk, 0, stream>>>(h_b, w2_t, b2, nullptr, nullptr, out,
                                           x, 2, 1, (int)Mrows, 768, 3072);
}
